// NanoTiledAttention_22883585753445
// MI455X (gfx1250) — compile-verified, matrix operand order checked by reading, not run on hardware
//
#include <hip/hip_runtime.h>

typedef _Float16 v8h  __attribute__((ext_vector_type(8)));
typedef _Float16 v16h __attribute__((ext_vector_type(16)));
typedef float    v8f  __attribute__((ext_vector_type(8)));

constexpr int Bc = 2, Lc = 2048, Ec = 1024, Hc = 16, Dc = 64;
constexpr int Mrows = Bc * Lc;  // 4096

__device__ __forceinline__ v16h cat8(v8h lo, v8h hi) {
    return __builtin_shufflevector(lo, hi, 0,1,2,3,4,5,6,7,8,9,10,11,12,13,14,15);
}

// WMMA f16 A-matrix (16x32) / B-matrix (32x16) fragment loader for data that is
// contiguous along K with leading dimension `ld` (in halves). Works for global
// (global_load_b128) and LDS (ds_load_b128) pointers alike.
// Layout (ISA 7.12.2, 16-bit A 16x32): lanes 0-15 -> row = lane, K = {0..7,16..23};
// lanes 16-31 -> row = lane-16, K = {8..15,24..31}; pairs packed low-K-first.
__device__ __forceinline__ v16h frag_ld(const _Float16* base, int ld, int row0, int k0) {
    const int lane = threadIdx.x & 31;
    const int r  = row0 + (lane & 15);
    const int kh = k0 + ((lane >> 4) << 3);
    const _Float16* p = base + (size_t)r * ld + kh;
    v8h lo = *(const v8h*)(p);
    v8h hi = *(const v8h*)(p + 16);
    return cat8(lo, hi);
}

__device__ __forceinline__ v8f wmma16(v16h a, v16h b, v8f c) {
    return __builtin_amdgcn_wmma_f32_16x16x32_f16(false, a, false, b,
                                                  (short)0, c, false, false);
}

// CDNA5 async copy: 16B global -> LDS, tracked by ASYNCcnt (inline asm; the
// __builtin_amdgcn_global_load_async_* arity differs across toolchains).
__device__ __forceinline__ void async_cp16(unsigned lds_off, const _Float16* g) {
    asm volatile("global_load_async_to_lds_b128 %0, %1, off"
                 :: "v"(lds_off), "v"(g) : "memory");
}
template <int N>
__device__ __forceinline__ void wait_asynccnt() {
    asm volatile("s_wait_asynccnt %0" :: "n"(N) : "memory");
}

__global__ void f32_to_f16_kernel(const float* __restrict__ in,
                                  _Float16* __restrict__ out, int n) {
    int i = blockIdx.x * blockDim.x + threadIdx.x;
    const int stride = gridDim.x * blockDim.x;
    for (; i < n; i += stride) out[i] = (_Float16)in[i];
}

// NT GEMM: Y[m,n] = oscale * (sum_k A[m,k] * W[n,k] + bias[n]).
// M x N x K = 4096 x 1024 x 1024. Block = 4 waves, 64x64 tile.
// MODE 0: store f16 [B,H,L,D] (Q,K)   MODE 1: store f16 [B,H,D,L] (V transposed)
// MODE 2: store f32 row-major [M,E]   (final projection)
template <int MODE>
__global__ __launch_bounds__(128) void gemm_nt(const _Float16* __restrict__ A,
                                               const _Float16* __restrict__ W,
                                               const float* __restrict__ bias,
                                               void* __restrict__ outp,
                                               float oscale) {
    const int lane = threadIdx.x & 31;
    const int wave = threadIdx.x >> 5;
    const int m0 = blockIdx.x * 64 + wave * 16;
    const int n0 = blockIdx.y * 64;

    v8f acc[4] = {};
    for (int k0 = 0; k0 < Ec; k0 += 32) {
        __builtin_prefetch((const void*)(A + (size_t)(m0 + (lane & 15)) * Ec + k0 + 128), 0, 3);
        v16h a = frag_ld(A, Ec, m0, k0);
#pragma unroll
        for (int t = 0; t < 4; ++t) {
            v16h b = frag_ld(W, Ec, n0 + t * 16, k0);
            acc[t] = wmma16(a, b, acc[t]);
        }
    }

    // C/D layout: VGPR r, lane l -> (m = r + 8*(l>>4), n = l&15)
#pragma unroll
    for (int t = 0; t < 4; ++t) {
        const int col = n0 + t * 16 + (lane & 15);
        const float bv = bias[col];
#pragma unroll
        for (int r = 0; r < 8; ++r) {
            const int m = m0 + r + ((lane >> 4) << 3);
            const float v = (acc[t][r] + bv) * oscale;
            if (MODE == 2) {
                ((float*)outp)[(size_t)m * Ec + col] = v;
            } else {
                const int h = col / Dc, d = col % Dc;
                const int b = m / Lc, l = m % Lc;
                const size_t idx = (MODE == 0)
                    ? ((size_t)((b * Hc + h) * Lc + l) * Dc + d)
                    : ((size_t)((b * Hc + h) * Dc + d) * Lc + l);
                ((_Float16*)outp)[idx] = (_Float16)v;
            }
        }
    }
}

// Transposed flash attention, K/V double-buffered in LDS via async copies.
// grid (L/64, B*H), 4 waves/block, wave owns 16 queries; all waves share the
// 32-key K/V tiles staged in LDS (4x less global traffic than per-wave loads).
// S^T = K Q^T keeps queries on the N axis: per-query softmax reductions are
// in-lane + one xor-16 shuffle, and P^T lands natively in B-fragment layout
// for O^T = V^T P^T (no LDS transpose of P needed).
__global__ __launch_bounds__(128) void attn_kernel(const _Float16* __restrict__ Qh,
                                                   const _Float16* __restrict__ Kh,
                                                   const _Float16* __restrict__ Vth,
                                                   _Float16* __restrict__ Oh) {
    __shared__ alignas(16) _Float16 Kt[2][32 * Dc];  // [buf][key][d]   4 KB each
    __shared__ alignas(16) _Float16 Vt[2][Dc * 32];  // [buf][d][key]   4 KB each

    const int tid  = threadIdx.x;
    const int lane = tid & 31;
    const int wave = tid >> 5;
    const int bh = blockIdx.y;
    const int b = bh / Hc, h = bh % Hc;
    const int q0 = blockIdx.x * 64 + wave * 16;

    const _Float16* Qb = Qh  + (size_t)bh * Lc * Dc;   // pre-scaled by 1/sqrt(D)
    const _Float16* Kb = Kh  + (size_t)bh * Lc * Dc;
    const _Float16* Vb = Vth + (size_t)bh * Dc * Lc;

    // Issue the async fills for one 32-key tile pair: 128 threads x 2 x 16B per
    // tile. K tile rows are 128B (8 chunks); V tile rows are 64B (4 chunks).
    auto issue_tiles = [&](int buf, int kb) {
#pragma unroll
        for (int j = 0; j < 2; ++j) {
            const int c = tid * 2 + j;  // 0..255
            const int krow = c >> 3, kch = c & 7;
            async_cp16((unsigned)(unsigned long long)(&Kt[buf][krow * Dc + kch * 8]),
                       Kb + (size_t)(kb + krow) * Dc + kch * 8);
            const int vrow = c >> 2, vch = c & 3;
            async_cp16((unsigned)(unsigned long long)(&Vt[buf][vrow * 32 + vch * 8]),
                       Vb + (size_t)vrow * Lc + kb + vch * 8);
        }
    };

    // Q as B-matrix fragments: B[k=d, n=query] = Q[q0+n, d]
    const v16h qb0 = frag_ld(Qb, Dc, q0, 0);
    const v16h qb1 = frag_ld(Qb, Dc, q0, 32);

    v8f o[4] = {};                 // O^T: rows = d (4 tiles of 16), cols = query
    float m_run = -1e30f;          // per-lane = per-query (query = lane & 15)
    float l_run = 0.f;

    int buf = 0;
    issue_tiles(0, 0);
    for (int kb = 0; kb < Lc; kb += 32) {
        const int nxt = kb + 32;
        if (nxt < Lc) {                 // block-uniform branch
            issue_tiles(buf ^ 1, nxt);
            wait_asynccnt<4>();         // in-order: current buf's 4 ops are done
        } else {
            wait_asynccnt<0>();
        }
        __syncthreads();                // all waves' tile portions visible

        const _Float16* Kbuf = &Kt[buf][0];
        const _Float16* Vbuf = &Vt[buf][0];

        // S^T tiles from LDS: keys 0..15 -> s0, 16..31 -> s1 (key = M, query = N)
        v8f s0 = {}, s1 = {};
        s0 = wmma16(frag_ld(Kbuf, Dc, 0,  0),  qb0, s0);
        s0 = wmma16(frag_ld(Kbuf, Dc, 0,  32), qb1, s0);
        s1 = wmma16(frag_ld(Kbuf, Dc, 16, 0),  qb0, s1);
        s1 = wmma16(frag_ld(Kbuf, Dc, 16, 32), qb1, s1);

        // per-query max over 32 keys: 15 in-lane max + one xor-16 combine
        float tm = fmaxf(s0[0], s1[0]);
#pragma unroll
        for (int r = 1; r < 8; ++r) tm = fmaxf(tm, fmaxf(s0[r], s1[r]));
        tm = fmaxf(tm, __shfl_xor(tm, 16, 32));

        const float mn   = fmaxf(m_run, tm);
        const float corr = __expf(m_run - mn);

        // P^T packed directly into the B-fragment for V^T x P^T:
        // frag halves 0..7 = tile0 keys r=0..7, halves 8..15 = tile1 keys r=0..7
        v16h pf;
        float rs = 0.f;
#pragma unroll
        for (int r = 0; r < 8; ++r) {
            const float p0 = __expf(s0[r] - mn);
            const float p1 = __expf(s1[r] - mn);
            rs += p0 + p1;
            pf[r]     = (_Float16)p0;
            pf[r + 8] = (_Float16)p1;
        }
        rs += __shfl_xor(rs, 16, 32);
        l_run = l_run * corr + rs;
        m_run = mn;

#pragma unroll
        for (int t = 0; t < 4; ++t) {
#pragma unroll
            for (int r = 0; r < 8; ++r) o[t][r] *= corr;
        }

        // O^T += V^T P^T : A = V^T rows d (contiguous along keys in LDS tile)
#pragma unroll
        for (int t = 0; t < 4; ++t) {
            v16h vf = frag_ld(Vbuf, 32, t * 16, 0);
            o[t] = wmma16(vf, pf, o[t]);
        }

        __syncthreads();                // reads done before buf is refilled
        buf ^= 1;
    }

    // finalize: per lane, o[t][r] -> d = t*16 + 8*(lane>>4) + r, query = lane&15.
    // 8 consecutive d-columns per tile -> one 16B v8h store per tile.
    const float inv = 1.f / l_run;
    const size_t row = (size_t)(b * Lc + q0 + (lane & 15)) * Ec;
#pragma unroll
    for (int t = 0; t < 4; ++t) {
        v8h ov;
#pragma unroll
        for (int r = 0; r < 8; ++r) ov[r] = (_Float16)(o[t][r] * inv);
        const int col = h * Dc + t * 16 + ((lane >> 4) << 3);
        *(v8h*)(Oh + row + col) = ov;
    }
}

extern "C" void kernel_launch(void* const* d_in, const int* in_sizes, int n_in,
                              void* d_out, int out_size, void* d_ws, size_t ws_size,
                              hipStream_t stream) {
    const float* X  = (const float*)d_in[0];
    const float* Wq = (const float*)d_in[1];
    const float* bq = (const float*)d_in[2];
    const float* Wk = (const float*)d_in[3];
    const float* bk = (const float*)d_in[4];
    const float* Wv = (const float*)d_in[5];
    const float* bv = (const float*)d_in[6];
    const float* Wo = (const float*)d_in[7];
    const float* bo = (const float*)d_in[8];

    char* ws = (char*)d_ws;
    _Float16* Xh  = (_Float16*)(ws);                          // 8 MB
    _Float16* Wqh = (_Float16*)(ws + (size_t)(8u  << 20));    // 2 MB each
    _Float16* Wkh = (_Float16*)(ws + (size_t)(10u << 20));
    _Float16* Wvh = (_Float16*)(ws + (size_t)(12u << 20));
    _Float16* Woh = (_Float16*)(ws + (size_t)(14u << 20));
    _Float16* Qh  = (_Float16*)(ws + (size_t)(16u << 20));    // 8 MB [B,H,L,D]
    _Float16* Kh  = (_Float16*)(ws + (size_t)(24u << 20));    // 8 MB [B,H,L,D]
    _Float16* Vth = (_Float16*)(ws + (size_t)(32u << 20));    // 8 MB [B,H,D,L]
    _Float16* Oh  = (_Float16*)(ws + (size_t)(40u << 20));    // 8 MB [B*L,E]

    const int nX = Mrows * Ec;  // 4,194,304
    const int nW = Ec * Ec;     // 1,048,576

    f32_to_f16_kernel<<<2048, 256, 0, stream>>>(X,  Xh,  nX);
    f32_to_f16_kernel<<<1024, 256, 0, stream>>>(Wq, Wqh, nW);
    f32_to_f16_kernel<<<1024, 256, 0, stream>>>(Wk, Wkh, nW);
    f32_to_f16_kernel<<<1024, 256, 0, stream>>>(Wv, Wvh, nW);
    f32_to_f16_kernel<<<1024, 256, 0, stream>>>(Wo, Woh, nW);

    const float qscale = 0.125f;  // 1/sqrt(64), folded into Q projection
    dim3 gg(Mrows / 64, Ec / 64); // 64 x 16
    gemm_nt<0><<<gg, 128, 0, stream>>>(Xh, Wqh, bq, (void*)Qh,  qscale);
    gemm_nt<0><<<gg, 128, 0, stream>>>(Xh, Wkh, bk, (void*)Kh,  1.0f);
    gemm_nt<1><<<gg, 128, 0, stream>>>(Xh, Wvh, bv, (void*)Vth, 1.0f);

    dim3 ga(Lc / 64, Bc * Hc);    // 32 x 32
    attn_kernel<<<ga, 128, 0, stream>>>(Qh, Kh, Vth, Oh);

    gemm_nt<2><<<gg, 128, 0, stream>>>(Oh, Woh, bo, d_out, 1.0f);
}
